// MAMBASR_25632364823023
// MI455X (gfx1250) — matrix operand order repaired and checked by reading, not compiled, not run on hardware
//
#include <hip/hip_runtime.h>
#include <hip/hip_bf16.h>
#include <math.h>

typedef __attribute__((ext_vector_type(16))) _Float16 v16h;
typedef __attribute__((ext_vector_type(8)))  _Float16 v8h;
typedef __attribute__((ext_vector_type(8)))  float    v8f;

// Problem constants (from reference setup_inputs)
#define Bn   4
#define Cc   64
#define Hh   64
#define Ww   64
#define HQ   256
#define WQ   256
#define INC  266          // real input channels of layer 0
#define KP1  288          // padded K for GEMM1 (9 * 32)
#define WID  256          // hidden width
#define TM   128          // rows (query pixels) per block
#define NB   4            // N-tiles register-blocked per wave
#define PIX_PER_IMG (HQ*WQ)          // 65536
#define BLK_PER_IMG (PIX_PER_IMG/TM) // 512

// d_ws layout (fp16, WMMA-fragment swizzled):
//   B1sw: 9 Ktiles * 16 Ntiles * 512 halves  = 73728 halves (w00^T padded)
//   B2sw: 8 Ktiles * 16 Ntiles * 512 halves  = 65536 halves (w1^T)
#define B1SW_HALVES (9*16*512)
#define B2SW_HALVES (8*16*512)

// ---------------------------------------------------------------------------
// Weight conversion: f32 row-major -> fp16 fragment-swizzled tiles.
// Tile (kt,nt) is 32x16; half index within tile: lane L (0..31), i (0..15)
//   k_local = (L>>4)*16 + i , n_local = L & 15
// so a lane's 16-half B fragment is 32 contiguous bytes.
// ---------------------------------------------------------------------------
__global__ __launch_bounds__(256) void liif_convert_weights(
    const float* __restrict__ w00, const float* __restrict__ w1,
    _Float16* __restrict__ ws)
{
  int e = blockIdx.x * blockDim.x + threadIdx.x;
  int total = B1SW_HALVES + B2SW_HALVES;
  if (e >= total) return;
  if (e < B1SW_HALVES) {
    int tile = e >> 9, within = e & 511;
    int lane = within >> 4, i = within & 15;
    int kt = tile >> 4, nt = tile & 15;
    int k = kt * 32 + ((lane >> 4) * 16 + i);
    int n = nt * 16 + (lane & 15);
    float v = (k < INC) ? w00[n * INC + k] : 0.0f;
    ws[e] = (_Float16)v;
  } else {
    int e2 = e - B1SW_HALVES;
    int tile = e2 >> 9, within = e2 & 511;
    int lane = within >> 4, i = within & 15;
    int kt = tile >> 4, nt = tile & 15;
    int k = kt * 32 + ((lane >> 4) * 16 + i);
    int n = nt * 16 + (lane & 15);
    ws[e] = (_Float16)w1[n * WID + k];
  }
}

__device__ __forceinline__ float clampf(float x, float lo, float hi) {
  return fminf(fmaxf(x, lo), hi);
}

__device__ __forceinline__ v16h load_frag_lds(const _Float16* p) {
  v8h a0 = *(const v8h*)(p);
  v8h a1 = *(const v8h*)(p + 8);
  return __builtin_shufflevector(a0, a1, 0,1,2,3,4,5,6,7,8,9,10,11,12,13,14,15);
}
__device__ __forceinline__ v16h load_frag_g(const _Float16* p) {
  v8h b0 = *(const v8h*)(p);
  v8h b1 = *(const v8h*)(p + 8);
  return __builtin_shufflevector(b0, b1, 0,1,2,3,4,5,6,7,8,9,10,11,12,13,14,15);
}

// ---------------------------------------------------------------------------
// Fused LIIF decoder. One block = 128 query pixels of one image.
// LDS: A1 grid [128][288] f16 | A2 x0 [128][256] f16 | meta | biases
// ---------------------------------------------------------------------------
__global__ __launch_bounds__(256, 2) void liif_fused_mlp(
    const float* __restrict__ inp,   // [B,3,64,64]
    const float* __restrict__ feat,  // [B,64,64,64]
    const float* __restrict__ coord, // [B,256,256,2]
    const float* __restrict__ cell,  // [B,2]
    const float* __restrict__ b00,   // [256]
    const float* __restrict__ b1,    // [256]
    const float* __restrict__ w2,    // [3,256]
    const float* __restrict__ b2,    // [3]
    const _Float16* __restrict__ ws, // swizzled B1|B2
    float* __restrict__ out)         // [B,3,256,256]
{
  extern __shared__ unsigned char smem[];
  _Float16* A1   = (_Float16*)smem;                 // 128*288 halves (73728 B)
  _Float16* A2   = A1 + TM * KP1;                   // 128*256 halves (65536 B)
  float*    wgt  = (float*)(A2 + TM * WID);         // 128*4
  int*      midx = (int*)(wgt + TM * 4);            // 128*4
  float*    cxy  = (float*)(midx + TM * 4);         // 128*2
  float*    bb00 = cxy + TM * 2;                    // 256
  float*    bb1  = bb00 + WID;                      // 256
  _Float16* Hl   = A1;                              // phase-2 output reuses A1

  const int tid  = threadIdx.x;
  const int lane = tid & 31;
  const int wave = tid >> 5;
  const int bimg = blockIdx.x / BLK_PER_IMG;        // batch index (whole block)
  const int q0   = blockIdx.x * TM;                 // first global pixel id

  // stage biases in LDS
  bb00[tid] = b00[tid];
  bb1[tid]  = b1[tid];

  // ------------------ Phase 0a: per-pixel coords/areas/rel channels ---------
  if (tid < TM) {
    const int m   = tid;
    const int q   = q0 + m;
    const int rem = q & (PIX_PER_IMG - 1);
    const int qy  = rem >> 8, qx = rem & 255;
    const float cy = coord[(((size_t)bimg * HQ + qy) * WQ + qx) * 2 + 0];
    const float cx = coord[(((size_t)bimg * HQ + qy) * WQ + qx) * 2 + 1];
    cxy[2 * m + 0] = cy; cxy[2 * m + 1] = cx;
    const float rx = 1.0f / Hh, ry = 1.0f / Ww, eps = 1e-6f;
    float area[4];
    #pragma unroll
    for (int c = 0; c < 4; ++c) {
      const float vx = (c < 2) ? -1.0f : 1.0f;
      const float vy = (c & 1) ? 1.0f : -1.0f;
      float y = clampf(cy + vx * rx + eps, -1.0f + 1e-6f, 1.0f - 1e-6f);
      float x = clampf(cx + vy * ry + eps, -1.0f + 1e-6f, 1.0f - 1e-6f);
      int iy = (int)clampf(rintf((y + 1.0f) * (Hh * 0.5f) - 0.5f), 0.0f, Hh - 1.0f);
      int ix = (int)clampf(rintf((x + 1.0f) * (Ww * 0.5f) - 0.5f), 0.0f, Ww - 1.0f);
      float oy = -1.0f + (2.0f * iy + 1.0f) / Hh;
      float ox = -1.0f + (2.0f * ix + 1.0f) / Ww;
      float r0 = (cy - oy) * Hh, r1 = (cx - ox) * Ww;
      area[c] = fabsf(r0 * r1) + 1e-9f;
      midx[m * 4 + c] = iy * Ww + ix;
      A1[m * KP1 + 2 * c + 0] = (_Float16)r0;
      A1[m * KP1 + 2 * c + 1] = (_Float16)r1;
    }
    const float tot = area[0] + area[1] + area[2] + area[3];
    #pragma unroll
    for (int c = 0; c < 4; ++c) wgt[m * 4 + c] = area[3 - c] / tot;  // corner swap
    A1[m * KP1 + 264] = (_Float16)(cell[bimg * 2 + 0] * (float)Hh);
    A1[m * KP1 + 265] = (_Float16)(cell[bimg * 2 + 1] * (float)Ww);
    #pragma unroll
    for (int k = INC; k < KP1; ++k) A1[m * KP1 + k] = (_Float16)0.0f;  // zero pad
  }
  __syncthreads();

  // ------------------ Phase 0b: area-weighted feature gathers --------------
  // 512 (pixel,corner) tasks over 256 threads; feat lives in L2 (4 MB << 192 MB)
  for (int t = tid; t < TM * 4; t += 256) {
    const int m = t >> 2, c = t & 3;
    const float w = wgt[m * 4 + c];
    const int idx = midx[m * 4 + c];
    const float* fp = feat + ((size_t)bimg * Cc) * (Hh * Ww) + idx;
    _Float16* dst = A1 + m * KP1 + 8 + c * Cc;
    #pragma unroll 8
    for (int ch = 0; ch < Cc; ++ch)
      dst[ch] = (_Float16)(fp[(size_t)ch * (Hh * Ww)] * w);
  }
  __syncthreads();

  const int khalf = lane >> 4;                  // which half of K this lane covers
  const int mrow0 = (wave << 4) + (lane & 15);  // A-frag row for this lane
  const int nlane = lane & 15;

  // ------------------ Phase 1: x0 = grid(288) x W00sw -> A2 (fp16) ---------
  // Register blocking: 4 N-tiles per pass -> A frag reused 4x, 4 independent
  // accumulator chains (no WMMA->WMMA RAW stalls).
  #pragma unroll 1
  for (int j = 0; j < 16 / NB; ++j) {
    v8f acc[NB] = {};
    const _Float16* bbase = ws + (size_t)(j * NB) * 512 + (size_t)lane * 16;
    #pragma unroll
    for (int kt = 0; kt < 9; ++kt) {
      v16h a = load_frag_lds(A1 + (size_t)mrow0 * KP1 + kt * 32 + khalf * 16);
      const _Float16* bp = bbase + (size_t)kt * (16 * 512);
      #pragma unroll
      for (int u = 0; u < NB; ++u) {
        v16h b = load_frag_g(bp + (size_t)u * 512);
        acc[u] = __builtin_amdgcn_wmma_f32_16x16x32_f16(false, a, false, b,
                                                        (short)0, acc[u], false, false);
      }
    }
    #pragma unroll
    for (int u = 0; u < NB; ++u) {
      #pragma unroll
      for (int r = 0; r < 8; ++r) {
        const int mr = (wave << 4) + r + (khalf << 3);
        const int n  = ((j * NB + u) << 4) + nlane;
        A2[mr * WID + n] = (_Float16)(acc[u][r] + bb00[n]);
      }
    }
  }
  __syncthreads();

  // ------------------ Phase 2: h = GELU(x0 x W1sw + b1) -> Hl (fp16) -------
  const _Float16* B2 = ws + B1SW_HALVES;
  #pragma unroll 1
  for (int j = 0; j < 16 / NB; ++j) {
    v8f acc[NB] = {};
    const _Float16* bbase = B2 + (size_t)(j * NB) * 512 + (size_t)lane * 16;
    #pragma unroll
    for (int kt = 0; kt < 8; ++kt) {
      v16h a = load_frag_lds(A2 + (size_t)mrow0 * WID + kt * 32 + khalf * 16);
      const _Float16* bp = bbase + (size_t)kt * (16 * 512);
      #pragma unroll
      for (int u = 0; u < NB; ++u) {
        v16h b = load_frag_g(bp + (size_t)u * 512);
        acc[u] = __builtin_amdgcn_wmma_f32_16x16x32_f16(false, a, false, b,
                                                        (short)0, acc[u], false, false);
      }
    }
    #pragma unroll
    for (int u = 0; u < NB; ++u) {
      #pragma unroll
      for (int r = 0; r < 8; ++r) {
        const int mr = (wave << 4) + r + (khalf << 3);
        const int n  = ((j * NB + u) << 4) + nlane;
        float x = acc[u][r] + bb1[n];
        float g = 0.5f * x * (1.0f + erff(x * 0.70710678f));   // exact GELU
        Hl[mr * WID + n] = (_Float16)g;
      }
    }
  }
  __syncthreads();

  // ------------------ Phase 3: out = h x w2^T + b2 + bilinear(inp) ---------
  for (int t = tid; t < TM * 3; t += 256) {
    const int m = t & (TM - 1), o = t >> 7;   // o in 0..2
    float acc = b2[o];
    const _Float16* hp = Hl + m * WID;
    const float* wp = w2 + o * WID;
    #pragma unroll 8
    for (int k = 0; k < WID; ++k) acc += (float)hp[k] * wp[k];
    // bilinear, padding_mode='border', align_corners=False
    const float cy = cxy[2 * m + 0], cx = cxy[2 * m + 1];
    float py = clampf((cy + 1.0f) * (Hh * 0.5f) - 0.5f, 0.0f, Hh - 1.0f);
    float px = clampf((cx + 1.0f) * (Ww * 0.5f) - 0.5f, 0.0f, Ww - 1.0f);
    int y0i = (int)floorf(py), x0i = (int)floorf(px);
    int y1i = min(y0i + 1, Hh - 1), x1i = min(x0i + 1, Ww - 1);
    float wy = py - y0i, wx = px - x0i;
    const float* ip = inp + ((size_t)bimg * 3 + o) * (Hh * Ww);
    float v00 = ip[y0i * Ww + x0i], v01 = ip[y0i * Ww + x1i];
    float v10 = ip[y1i * Ww + x0i], v11 = ip[y1i * Ww + x1i];
    float bil = v00 * (1 - wy) * (1 - wx) + v01 * (1 - wy) * wx
              + v10 * wy * (1 - wx)       + v11 * wy * wx;
    const int q = q0 + m, rem = q & (PIX_PER_IMG - 1);
    const int qy = rem >> 8, qx = rem & 255;
    out[(((size_t)bimg * 3 + o) << 16) + (qy << 8) + qx] = acc + bil;
  }
}

extern "C" void kernel_launch(void* const* d_in, const int* in_sizes, int n_in,
                              void* d_out, int out_size, void* d_ws, size_t ws_size,
                              hipStream_t stream) {
  const float* inp   = (const float*)d_in[0];
  const float* feat  = (const float*)d_in[1];
  const float* coord = (const float*)d_in[2];
  const float* cell  = (const float*)d_in[3];
  const float* w00   = (const float*)d_in[4];
  const float* b00   = (const float*)d_in[5];
  const float* w1    = (const float*)d_in[6];
  const float* b1    = (const float*)d_in[7];
  const float* w2    = (const float*)d_in[8];
  const float* b2    = (const float*)d_in[9];
  float* out = (float*)d_out;
  _Float16* ws = (_Float16*)d_ws;   // needs (73728+65536)*2 = 278528 bytes

  // 1) convert + fragment-swizzle weights into workspace (fp16)
  {
    int total = B1SW_HALVES + B2SW_HALVES;
    liif_convert_weights<<<(total + 255) / 256, 256, 0, stream>>>(w00, w1, ws);
  }
  // 2) fused gather + WMMA MLP + residual
  {
    const int nblocks = (Bn * PIX_PER_IMG) / TM;   // 2048
    const size_t smem = (size_t)TM * KP1 * 2 + (size_t)TM * WID * 2
                      + TM * 4 * 4 + TM * 4 * 4 + TM * 2 * 4 + WID * 4 + WID * 4;
    liif_fused_mlp<<<nblocks, 256, smem, stream>>>(inp, feat, coord, cell,
                                                   b00, b1, w2, b2, ws, out);
  }
}
